// Conv2d_82162724373128
// MI455X (gfx1250) — matrix operand order repaired and checked by reading, not compiled, not run on hardware
//
#include <hip/hip_runtime.h>
#include <stdint.h>

#define IN_CH   128
#define OUT_CH  256
#define HW      56
#define NBATCH  32

#define TH      8            // output rows per workgroup tile
#define TW      16           // output cols per workgroup tile
#define CCHUNK  32           // in-channels per K-chunk (one bf16 WMMA K)
#define XROWS   (TH + 2)     // 10 (halo)
#define XCOLS   (TW + 2)     // 18 (halo)
#define CPAD    40           // padded channel stride (80B = 20 dwords -> conflict-free)

#define WSLAB   4096                  // 128 outch * 32 c (ushorts) per precision plane
#define WS_NEEDED ((size_t)2 * 4 * 9 * 2 * WSLAB * sizeof(uint16_t))  // kb*cc*t*p planes

// CDNA5 async global->LDS path (ASYNCcnt), guarded so the file compiles on
// toolchains that lack the builtins (falls back to load+ds_store staging).
#if defined(__HIP_DEVICE_COMPILE__) &&                                         \
    __has_builtin(__builtin_amdgcn_global_load_async_to_lds_b128) &&           \
    __has_builtin(__builtin_amdgcn_s_wait_asynccnt)
#define HAVE_ASYNC_LDS 1
#define AS_GLOBAL __attribute__((address_space(1)))
#define AS_LOCAL  __attribute__((address_space(3)))
#else
#define HAVE_ASYNC_LDS 0
#endif

typedef __attribute__((ext_vector_type(16))) __bf16        v16bf;
typedef __attribute__((ext_vector_type(8)))  float         v8f;
typedef __attribute__((ext_vector_type(4)))  unsigned int  u32x4;
typedef __attribute__((ext_vector_type(4)))  int           v4i;

union Frag { u32x4 q[2]; v16bf v; };

__device__ __forceinline__ uint16_t f32_to_bf16(float f) {
    uint32_t u = __builtin_bit_cast(uint32_t, f);
    uint32_t r = u + 0x7FFFu + ((u >> 16) & 1u);   // round-to-nearest-even
    return (uint16_t)(r >> 16);
}
__device__ __forceinline__ float bf16_to_f32(uint16_t h) {
    uint32_t u = ((uint32_t)h) << 16;
    return __builtin_bit_cast(float, u);
}

// A-matrix 16x32 bf16 layout: lane<16 holds K {0..7,16..23}, lane>=16 holds {8..15,24..31}.
// Store weights permuted so each lane's 16 values are contiguous: swap bits 3<->4 of c.
__device__ __forceinline__ int permC(int c) {
    return (c & 7) | ((c & 8) << 1) | ((c & 16) >> 1);
}

// ---------------------------------------------------------------------------
// Prep: fp32 OIHW weights -> bf16 hi/lo planes, pre-permuted into the exact
// LDS A-fragment order. Layout: slab s = ((kb*4 + cc)*9 + t), each slab =
// [p(hi/lo)][o:128][permC(c):32] ushorts; per-tap staging = one 16 KB memcpy.
// ---------------------------------------------------------------------------
__global__ void __launch_bounds__(256)
prep_weights(const float* __restrict__ w, uint16_t* __restrict__ whl) {
    int i = blockIdx.x * 256 + threadIdx.x;     // flat (ko, c, t)
    if (i >= OUT_CH * IN_CH * 9) return;
    float v = w[i];                              // ((ko*128+c)*9 + t) == i
    uint16_t hi = f32_to_bf16(v);
    uint16_t lo = f32_to_bf16(v - bf16_to_f32(hi));
    int t  = i % 9;
    int c  = (i / 9) % IN_CH;
    int ko = i / (9 * IN_CH);
    int kb = ko >> 7, o = ko & 127, cc = c >> 5, ci = c & 31;
    size_t base = (size_t)(((kb * 4 + cc) * 9 + t) * 2) * WSLAB;
    int off = o * 32 + permC(ci);
    whl[base + off]         = hi;
    whl[base + WSLAB + off] = lo;
}

// ---------------------------------------------------------------------------
// Main conv: implicit GEMM, bf16x3 split precision, double-buffered W slabs.
// ---------------------------------------------------------------------------
template <bool PREPPED>
__global__ void __launch_bounds__(256)
conv3x3_wmma(const float* __restrict__ x, const float* __restrict__ w,
             const uint16_t* __restrict__ whl, float* __restrict__ out) {
    __shared__ uint16_t sXhi[XROWS * XCOLS * CPAD];
    __shared__ uint16_t sXlo[XROWS * XCOLS * CPAD];
    __shared__ uint16_t sW[2][2][128 * CPAD];        // [buf][hi/lo][o][cpad]

    const int bid = blockIdx.x;
    const int wt  = bid & 3;                 // 4 w-tiles (last overlaps past 56)
    const int ht  = (bid >> 2) % 7;          // 7 h-tiles
    const int kb  = (bid / 28) & 1;          // 2 outch blocks of 128
    const int n   = bid / 56;                // batch

    const int h0  = ht * TH;
    const int w0  = wt * TW;
    const int ko0 = kb * 128;

    const int tid  = threadIdx.x;
    const int lane = tid & 31;
    const int wid  = tid >> 5;
    const int wm   = wid & 1;                // 2 waves along M (64 outch each)
    const int wn   = wid >> 1;               // 4 waves along N (2 rows each)
    const int lrow = lane & 15;
    const int hsel = (lane < 16) ? 0 : 16;   // half-wave K offset

    v8f acc[4][2];
#pragma unroll
    for (int m = 0; m < 4; ++m)
#pragma unroll
        for (int nn = 0; nn < 2; ++nn) acc[m][nn] = (v8f)(0.0f);

    // Stage one 128x32 hi+lo weight slab for tap t into LDS buffer `buf`.
    auto stageW = [&](int t, int buf, int c0) {
        if (PREPPED) {
            const uint16_t* src =
                whl + (size_t)(((kb * 4 + (c0 >> 5)) * 9 + t) * 2) * WSLAB;
            int u0 = tid * 32;               // 64 bytes per thread, fully coalesced
            if (t < 8) __builtin_prefetch(src + 2 * WSLAB, 0, 0);
#pragma unroll
            for (int k = 0; k < 4; ++k) {
                int u   = u0 + k * 8;
                int p   = u >> 12;
                int rem = u & (WSLAB - 1);
                int o   = rem >> 5;
                int ci  = rem & 31;          // already permuted, 8-ushort segments
#if HAVE_ASYNC_LDS
                // DMA-style copy on the ASYNCcnt path: no VGPR round trip.
                __builtin_amdgcn_global_load_async_to_lds_b128(
                    (AS_GLOBAL v4i*)(src + u),
                    (AS_LOCAL v4i*)&sW[buf][p][o * CPAD + ci], 0, 0);
#else
                u32x4 d = *(const u32x4*)(src + u);
                *(u32x4*)(&sW[buf][p][o * CPAD + ci]) = d;
#endif
            }
        } else {
            int kh = t / 3, kw = t - kh * 3;
            for (int idx = tid; idx < 128 * CCHUNK; idx += 256) {
                int o = idx >> 5;
                int c = idx & 31;
                float v = w[(((size_t)(ko0 + o) * IN_CH + (c0 + c)) * 3 + kh) * 3 + kw];
                uint16_t hi = f32_to_bf16(v);
                uint16_t lo = f32_to_bf16(v - bf16_to_f32(hi));
                int l = o * CPAD + permC(c);
                sW[buf][0][l] = hi;
                sW[buf][1][l] = lo;
            }
        }
    };

    // Drain async staging before a barrier publishes the slab.
    auto drainAsync = [&]() {
#if HAVE_ASYNC_LDS
        if (PREPPED) __builtin_amdgcn_s_wait_asynccnt(0);
#endif
    };

    for (int c0 = 0; c0 < IN_CH; c0 += CCHUNK) {
        // (previous chunk's tap-8 __syncthreads protects sX / sW[0] reuse)

        // ---- stage input tile (with halo) as bf16 hi/lo, layout [r][col][c] ----
        for (int idx = tid; idx < CCHUNK * XROWS * XCOLS; idx += 256) {
            int c   = idx / (XROWS * XCOLS);
            int rem = idx - c * (XROWS * XCOLS);
            int r   = rem / XCOLS;
            int col = rem - r * XCOLS;
            int gh  = h0 - 1 + r;
            int gw  = w0 - 1 + col;
            float v = 0.0f;
            if (gh >= 0 && gh < HW && gw >= 0 && gw < HW) {
                size_t gi = (((size_t)n * IN_CH + (c0 + c)) * HW + gh) * HW + gw;
                v = x[gi];
                if (c0 + CCHUNK < IN_CH)
                    __builtin_prefetch(&x[gi + (size_t)CCHUNK * HW * HW], 0, 0);
            }
            uint16_t hi = f32_to_bf16(v);
            uint16_t lo = f32_to_bf16(v - bf16_to_f32(hi));
            int l = (r * XCOLS + col) * CPAD + permC(c);
            sXhi[l] = hi;
            sXlo[l] = lo;
        }
        stageW(0, 0, c0);
        drainAsync();
        __syncthreads();

        for (int t = 0; t < 9; ++t) {
            const int buf = t & 1;
            const int kh  = t / 3;
            const int kw  = t - kh * 3;

            // ---- fragment loads first (WMMA waits only on these) ----
            Frag Ahi[4], Alo[4];
#pragma unroll
            for (int m = 0; m < 4; ++m) {
                int off = (wm * 64 + m * 16 + lrow) * CPAD + hsel;
                Ahi[m].q[0] = *(const u32x4*)(&sW[buf][0][off]);
                Ahi[m].q[1] = *(const u32x4*)(&sW[buf][0][off + 8]);
                Alo[m].q[0] = *(const u32x4*)(&sW[buf][1][off]);
                Alo[m].q[1] = *(const u32x4*)(&sW[buf][1][off + 8]);
            }
            Frag Bhi[2], Blo[2];
#pragma unroll
            for (int nn = 0; nn < 2; ++nn) {
                int r   = wn * 2 + nn + kh;
                int col = lrow + kw;
                int off = (r * XCOLS + col) * CPAD + hsel;
                Bhi[nn].q[0] = *(const u32x4*)(sXhi + off);
                Bhi[nn].q[1] = *(const u32x4*)(sXhi + off + 8);
                Blo[nn].q[0] = *(const u32x4*)(sXlo + off);
                Blo[nn].q[1] = *(const u32x4*)(sXlo + off + 8);
            }

            // ---- overlap: stage next tap's W into the other buffer ----
            if (t < 8) stageW(t + 1, buf ^ 1, c0);

            // ---- bf16x3: hi*hi + hi*lo + lo*hi, f32 accumulate ----
#pragma unroll
            for (int m = 0; m < 4; ++m) {
#pragma unroll
                for (int nn = 0; nn < 2; ++nn) {
                    acc[m][nn] = __builtin_amdgcn_wmma_f32_16x16x32_bf16(
                        false, Ahi[m].v, false, Bhi[nn].v, (short)0, acc[m][nn], false, false);
                    acc[m][nn] = __builtin_amdgcn_wmma_f32_16x16x32_bf16(
                        false, Ahi[m].v, false, Blo[nn].v, (short)0, acc[m][nn], false, false);
                    acc[m][nn] = __builtin_amdgcn_wmma_f32_16x16x32_bf16(
                        false, Alo[m].v, false, Bhi[nn].v, (short)0, acc[m][nn], false, false);
                }
            }
            drainAsync();      // async staging of t+1 complete...
            __syncthreads();   // ...and published; readers of buf done
        }
    }

    // ---- store: D vgpr i -> M = i + (lane>=16 ? 8:0), N = lane%16 ----
    const int ww = w0 + lrow;
    if (ww < HW) {
#pragma unroll
        for (int m = 0; m < 4; ++m) {
#pragma unroll
            for (int nn = 0; nn < 2; ++nn) {
                int hh = h0 + wn * 2 + nn;
#pragma unroll
                for (int i = 0; i < 8; ++i) {
                    int oc = ko0 + wm * 64 + m * 16 + i + ((lane < 16) ? 0 : 8);
                    out[(((size_t)n * OUT_CH + oc) * HW + hh) * HW + ww] = acc[m][nn][i];
                }
            }
        }
    }
}

extern "C" void kernel_launch(void* const* d_in, const int* in_sizes, int n_in,
                              void* d_out, int out_size, void* d_ws, size_t ws_size,
                              hipStream_t stream) {
    (void)in_sizes; (void)n_in; (void)out_size;
    const float* x = (const float*)d_in[0];
    const float* w = (const float*)d_in[1];
    float* out     = (float*)d_out;

    dim3 grid(NBATCH * 2 * 7 * 4);   // batch * outch-blocks * h-tiles * w-tiles
    dim3 block(256);

    if (d_ws && ws_size >= WS_NEEDED) {
        uint16_t* whl = (uint16_t*)d_ws;
        prep_weights<<<dim3((OUT_CH * IN_CH * 9 + 255) / 256), block, 0, stream>>>(w, whl);
        conv3x3_wmma<true><<<grid, block, 0, stream>>>(x, w, whl, out);
    } else {
        conv3x3_wmma<false><<<grid, block, 0, stream>>>(x, w, nullptr, out);
    }
}
